// SupernodePooling_18588618457315
// MI455X (gfx1250) — matrix operand order repaired and checked by reading, not compiled, not run on hardware
//
#include <hip/hip_runtime.h>
#include <hip/hip_bf16.h>
#include <math.h>

typedef __attribute__((ext_vector_type(16))) _Float16 v16h;
typedef __attribute__((ext_vector_type(8)))  _Float16 v8h;
typedef __attribute__((ext_vector_type(8)))  float    v8f;

#define N_PTS 4096
#define M_SUP 128
#define HID   192

// ---------------------------------------------------------------------------
// 1) Sincos embeddings for all points (pe) and supernodes (se), written as f16
//    for the WMMA GEMMs. Also gathers supernode positions.
//    c = d*64 + j ; j<32 -> sin(coord*omega_j), j>=32 -> cos(coord*omega_{j-32})
//    omega_j = 10000^(-j/32) = exp(-j * ln(10000)/32)
// ---------------------------------------------------------------------------
__global__ void sp_embed_kernel(const float* __restrict__ pos,
                                const int* __restrict__ sidx,
                                _Float16* __restrict__ peh,
                                _Float16* __restrict__ seh,
                                float* __restrict__ supPos) {
  int t = blockIdx.x * blockDim.x + threadIdx.x;
  const int total = (N_PTS + M_SUP) * HID;
  if (t >= total) return;
  int r = t / HID, c = t % HID;
  int d = c / 64, j = c % 64;
  bool is_cos = (j >= 32);
  int f = is_cos ? (j - 32) : j;
  float coord;
  _Float16* dst;
  if (r < N_PTS) {
    coord = pos[r * 3 + d];
    dst = peh + r * HID + c;
  } else {
    int m = r - N_PTS;
    int idx = sidx[m];
    coord = pos[idx * 3 + d];
    dst = seh + m * HID + c;
    if (c < 3) supPos[m * 3 + c] = pos[idx * 3 + c];
  }
  // ln(10000)/32
  float omega = __expf(-(float)f * 0.2878186906f);
  float ang = coord * omega;
  float v = is_cos ? __cosf(ang) : __sinf(ang);
  *dst = (_Float16)v;
}

// ---------------------------------------------------------------------------
// 2) Convert weights to f16 AND transpose them (Wt[n][k] = W[k][n]) so the
//    WMMA B operand (per-lane column n, consecutive K) becomes two aligned
//    16-byte loads instead of 16 strided u16 loads.
//    W1: 384x192 -> W1t: 192x384 ; W2: 192x192 -> W2t: 192x192 ;
//    Wp: 384x192 -> Wpt: 192x384
// ---------------------------------------------------------------------------
__global__ void sp_cvt_kernel(const float* __restrict__ W1,
                              const float* __restrict__ W2,
                              const float* __restrict__ Wp,
                              _Float16* __restrict__ W1t,
                              _Float16* __restrict__ W2t,
                              _Float16* __restrict__ Wpt) {
  int t = blockIdx.x * blockDim.x + threadIdx.x;
  const int s1 = 2 * HID * HID;   // 384*192
  const int s2 = HID * HID;       // 192*192
  const int s3 = 2 * HID * HID;   // 384*192
  if (t < s1) {
    int k = t / HID, n = t % HID;
    W1t[n * (2 * HID) + k] = (_Float16)W1[t];
  } else if (t < s1 + s2) {
    int u = t - s1;
    int k = u / HID, n = u % HID;
    W2t[n * HID + k] = (_Float16)W2[u];
  } else if (t < s1 + s2 + s3) {
    int u = t - s1 - s2;
    int k = u / HID, n = u % HID;
    Wpt[n * (2 * HID) + k] = (_Float16)Wp[u];
  }
}

// ---------------------------------------------------------------------------
// 3) Generic WMMA GEMM:  C(rows x 192) = A(rows x 192) @ W(192 x 192)
//    W supplied TRANSPOSED: Wt[n][k], row stride ldw, K-offset pre-applied by
//    the caller for the bottom halves of W1/Wp.
//    One wave (32 lanes) per 16x16 output tile; K loop = 6 x wmma 16x16x32.
//    Optional bias, optional f16 copy of result, optional accumulate (beta=1).
// ---------------------------------------------------------------------------
__global__ void sp_wmma_gemm(const _Float16* __restrict__ A,
                             const _Float16* __restrict__ Wt, int ldw,
                             const float* __restrict__ bias,
                             float* __restrict__ C,
                             _Float16* __restrict__ Ch,
                             int beta) {
  const int tiles_n = HID / 16;          // 12
  int tile_m = blockIdx.x / tiles_n;
  int tile_n = blockIdx.x % tiles_n;
  int lane = threadIdx.x;                // 0..31, one wave32 per block
  int mrow = lane & 15;
  int hi   = lane >> 4;                  // 0 or 1 (lane half)
  int koffA = hi * 8;                    // A layout: K quarters per lane half
  int koffB = hi * 8;                    // B layout: K quarters per lane half
  int ncol = lane & 15;

  int crow0 = tile_m * 16 + hi * 8;
  int ccol  = tile_n * 16 + ncol;

  v8f c = {};
  if (beta) {
#pragma unroll
    for (int r = 0; r < 8; ++r) c[r] = C[(size_t)(crow0 + r) * HID + ccol];
  }

  const _Float16* arow = A + (size_t)(tile_m * 16 + mrow) * HID;
  const _Float16* wrow = Wt + (size_t)(tile_n * 16 + ncol) * ldw;

#pragma unroll
  for (int kk = 0; kk < HID; kk += 32) {
    // A operand: lane half holds K = {koffA..koffA+7} and {16+koffA..+7}
    v8h a0 = *(const v8h*)(arow + kk + koffA);
    v8h a1 = *(const v8h*)(arow + kk + 16 + koffA);
    v16h a = __builtin_shufflevector(a0, a1, 0, 1, 2, 3, 4, 5, 6, 7,
                                            8, 9, 10, 11, 12, 13, 14, 15);
    // B operand: lane half holds K = {koffB..koffB+7} and {16+koffB..+7}
    v8h b0 = *(const v8h*)(wrow + kk + koffB);
    v8h b1 = *(const v8h*)(wrow + kk + 16 + koffB);
    v16h b = __builtin_shufflevector(b0, b1, 0, 1, 2, 3, 4, 5, 6, 7,
                                            8, 9, 10, 11, 12, 13, 14, 15);
    c = __builtin_amdgcn_wmma_f32_16x16x32_f16(
        /*neg_a=*/false, a, /*neg_b=*/false, b,
        /*c_mod=*/(short)0, c, /*reuse_a=*/false, /*reuse_b=*/false);
  }

#pragma unroll
  for (int r = 0; r < 8; ++r) {
    float v = c[r];
    if (bias) v += bias[ccol];
    C[(size_t)(crow0 + r) * HID + ccol] = v;
    if (Ch) Ch[(size_t)(crow0 + r) * HID + ccol] = (_Float16)v;
  }
}

// ---------------------------------------------------------------------------
// 4) Masked gelu accumulation. Block m (128 blocks), 192 threads = feature h.
//    Distance test identical across lanes -> wave-uniform branch, ~94% of
//    points skip the load+gelu entirely.
//    Sh[m,h] = (1/max(cnt,1)) * sum_{n: |p_n - s_m|^2 <= R^2}
//              gelu(A[n,h] + Bse[m,h] + b1[h])
// ---------------------------------------------------------------------------
__global__ void sp_mask_accum(const float* __restrict__ pos,
                              const float* __restrict__ supPos,
                              const float* __restrict__ Apre,
                              const float* __restrict__ Bse,
                              const float* __restrict__ b1,
                              _Float16* __restrict__ Sh) {
  int m = blockIdx.x;
  int h = threadIdx.x;                   // 0..191
  float base = Bse[m * HID + h] + b1[h];
  float sx = supPos[m * 3 + 0];
  float sy = supPos[m * 3 + 1];
  float sz = supPos[m * 3 + 2];
  float sum = 0.0f, cnt = 0.0f;
  for (int n = 0; n < N_PTS; ++n) {
    float dx = pos[n * 3 + 0] - sx;
    float dy = pos[n * 3 + 1] - sy;
    float dz = pos[n * 3 + 2] - sz;
    float sq = dx * dx + dy * dy + dz * dz;
    if (sq <= 0.0625f) {                 // RADIUS^2, uniform across wave
      float x = Apre[(size_t)n * HID + h] + base;
      sum += 0.5f * x * (1.0f + erff(x * 0.70710678118f));  // exact gelu
      cnt += 1.0f;
    }
  }
  float d = fmaxf(cnt, 1.0f);
  Sh[m * HID + h] = (_Float16)(sum / d);
}

// ---------------------------------------------------------------------------
// Launch
// ---------------------------------------------------------------------------
extern "C" void kernel_launch(void* const* d_in, const int* in_sizes, int n_in,
                              void* d_out, int out_size, void* d_ws, size_t ws_size,
                              hipStream_t stream) {
  (void)in_sizes; (void)n_in; (void)out_size; (void)ws_size;
  const float* pos  = (const float*)d_in[0];  // (4096,3)
  const float* W1   = (const float*)d_in[1];  // (384,192)
  const float* b1   = (const float*)d_in[2];  // (192)
  const float* W2   = (const float*)d_in[3];  // (192,192)
  const float* b2   = (const float*)d_in[4];  // (192)
  const float* Wp   = (const float*)d_in[5];  // (384,192)
  const float* bp   = (const float*)d_in[6];  // (192)
  const int*   sidx = (const int*)d_in[7];    // (128)
  float* out = (float*)d_out;                 // (128,192) f32

  // Workspace carve-up (256B aligned)
  char* ws = (char*)d_ws;
  size_t o = 0;
  auto carve = [&](size_t bytes) -> char* {
    char* p = ws + o;
    o = (o + bytes + 255) & ~(size_t)255;
    return p;
  };
  _Float16* peh    = (_Float16*)carve((size_t)N_PTS * HID * 2);
  _Float16* seh    = (_Float16*)carve((size_t)M_SUP * HID * 2);
  float*    supPos = (float*)   carve((size_t)M_SUP * 3 * 4);
  float*    Apre   = (float*)   carve((size_t)N_PTS * HID * 4);
  float*    Bse    = (float*)   carve((size_t)M_SUP * HID * 4);
  _Float16* Sh     = (_Float16*)carve((size_t)M_SUP * HID * 2);
  float*    aggf   = (float*)   carve((size_t)M_SUP * HID * 4);
  _Float16* aggh   = (_Float16*)carve((size_t)M_SUP * HID * 2);
  _Float16* W1t    = (_Float16*)carve((size_t)2 * HID * HID * 2);  // 192x384
  _Float16* W2t    = (_Float16*)carve((size_t)HID * HID * 2);      // 192x192
  _Float16* Wpt    = (_Float16*)carve((size_t)2 * HID * HID * 2);  // 192x384

  // 1) embeddings (points + supernodes) + supernode gather
  {
    int total = (N_PTS + M_SUP) * HID;
    sp_embed_kernel<<<(total + 255) / 256, 256, 0, stream>>>(pos, sidx, peh, seh, supPos);
  }
  // 2) weights -> f16, transposed
  {
    int total = (2 * HID + HID + 2 * HID) * HID;
    sp_cvt_kernel<<<(total + 255) / 256, 256, 0, stream>>>(W1, W2, Wp, W1t, W2t, Wpt);
  }
  const int TN = HID / 16;  // 12 column tiles
  // 3) A = pe @ W1_top          (4096x192); W1t rows are n, K offset 0
  sp_wmma_gemm<<<(N_PTS / 16) * TN, 32, 0, stream>>>(peh, W1t, 2 * HID, nullptr, Apre, nullptr, 0);
  // 4) Bse = se @ W1_bot        (128x192);  K offset +192 within W1t rows
  sp_wmma_gemm<<<(M_SUP / 16) * TN, 32, 0, stream>>>(seh, W1t + HID, 2 * HID, nullptr, Bse, nullptr, 0);
  // 5) masked gelu accumulation -> Sh = (masked mean of gelu) in f16
  sp_mask_accum<<<M_SUP, HID, 0, stream>>>(pos, supPos, Apre, Bse, b1, Sh);
  // 6) agg = Sh @ W2 + b2  (also f16 copy for final GEMM)
  sp_wmma_gemm<<<(M_SUP / 16) * TN, 32, 0, stream>>>(Sh, W2t, HID, b2, aggf, aggh, 0);
  // 7) out = se @ Wp_bot + bp
  sp_wmma_gemm<<<(M_SUP / 16) * TN, 32, 0, stream>>>(seh, Wpt + HID, 2 * HID, bp, out, nullptr, 0);
  // 8) out += agg @ Wp_top
  sp_wmma_gemm<<<(M_SUP / 16) * TN, 32, 0, stream>>>(aggh, Wpt, 2 * HID, nullptr, out, nullptr, 1);
}
